// PAConv_10943576670260
// MI455X (gfx1250) — matrix operand order repaired and checked by reading, not compiled, not run on hardware
//
#include <hip/hip_runtime.h>
#include <math.h>

typedef __attribute__((ext_vector_type(16))) _Float16 v16h;
typedef __attribute__((ext_vector_type(8)))  _Float16 v8h;
typedef __attribute__((ext_vector_type(8)))  float    v8f;

static constexpr int BB   = 32;
static constexpr int NN   = 1024;
static constexpr int RR   = BB * NN;      // 32768 points
static constexpr int KNN_ = 20;
static constexpr float EPS_ = 1e-5f;

// ---------------------------------------------------------------------------
// helpers
// ---------------------------------------------------------------------------
__device__ __forceinline__ v16h load_bfrag(const _Float16* __restrict__ Wsw,
                                           int fragIdx, int lane) {
  return *(const v16h*)(Wsw + ((size_t)fragIdx * 32 + lane) * 16);
}

// ---------------------------------------------------------------------------
// 0. transpose x (B,3,N) -> xT (R,3)
// ---------------------------------------------------------------------------
__global__ void k_transpose_x(const float* __restrict__ x, float* __restrict__ xT) {
  int t = blockIdx.x * blockDim.x + threadIdx.x;
  if (t >= RR * 3) return;
  int r = t / 3, c = t - r * 3;
  int b = r >> 10, n = r & 1023;
  xT[r * 3 + c] = x[(b * 3 + c) * NN + n];
}

// ---------------------------------------------------------------------------
// 1. KNN: per point, top-20 of pd = 2*inner - sq_n - sq_m (register insertion)
// ---------------------------------------------------------------------------
__global__ void k_knn(const float* __restrict__ x, int* __restrict__ idxOut) {
  int r = blockIdx.x * blockDim.x + threadIdx.x;
  if (r >= RR) return;
  int b = r >> 10, n = r & 1023;
  const float* xb = x + (size_t)b * 3 * NN;
  float q0 = xb[n], q1 = xb[NN + n], q2 = xb[2 * NN + n];
  float sqn = q0 * q0 + q1 * q1 + q2 * q2;
  float best[KNN_]; int bi[KNN_];
#pragma unroll
  for (int i = 0; i < KNN_; ++i) { best[i] = -3.4e38f; bi[i] = 0; }
  for (int m = 0; m < NN; ++m) {
    float a0 = xb[m], a1 = xb[NN + m], a2 = xb[2 * NN + m];
    float inner = q0 * a0 + q1 * a1 + q2 * a2;
    float sqm = a0 * a0 + a1 * a1 + a2 * a2;
    float pd = 2.0f * inner - sqn - sqm;
    if (pd > best[KNN_ - 1]) {
      best[KNN_ - 1] = pd; bi[KNN_ - 1] = m;
#pragma unroll
      for (int j = KNN_ - 1; j > 0; --j) {
        if (best[j] > best[j - 1]) {
          float tv = best[j]; best[j] = best[j - 1]; best[j - 1] = tv;
          int ti = bi[j]; bi[j] = bi[j - 1]; bi[j - 1] = ti;
        }
      }
    }
  }
#pragma unroll
  for (int j = 0; j < KNN_; ++j) idxOut[(size_t)r * KNN_ + j] = bi[j];
}

// ---------------------------------------------------------------------------
// 2. xyz (R,20,6) = [nx - center, nx]
// ---------------------------------------------------------------------------
__global__ void k_xyz(const float* __restrict__ xT, const int* __restrict__ idx,
                      float* __restrict__ xyz) {
  int t = blockIdx.x * blockDim.x + threadIdx.x;
  if (t >= RR * KNN_) return;
  int r = t / KNN_;
  int b = r >> 10;
  int nb = idx[t];
  const float* ctr = xT + (size_t)r * 3;
  const float* nx  = xT + ((size_t)(b * NN + nb)) * 3;
  float* o = xyz + (size_t)t * 6;
  o[0] = nx[0] - ctr[0]; o[1] = nx[1] - ctr[1]; o[2] = nx[2] - ctr[2];
  o[3] = nx[0];          o[4] = nx[1];          o[5] = nx[2];
}

// ---------------------------------------------------------------------------
// 3a. scorenet BN stats over h = xyz @ w1 (655360 rows x 16 ch); block/channel
// ---------------------------------------------------------------------------
__global__ __launch_bounds__(256) void k_sn_stats(const float* __restrict__ xyz,
                                                  const float* __restrict__ w1,
                                                  float* __restrict__ snst) {
  __shared__ float red[512];
  int ch = blockIdx.x, tid = threadIdx.x;
  float w[6];
#pragma unroll
  for (int i = 0; i < 6; ++i) w[i] = w1[i * 16 + ch];
  float s = 0.f, s2 = 0.f;
  const int ROWS = RR * KNN_;
  for (int r = tid; r < ROWS; r += 256) {
    const float* xr = xyz + (size_t)r * 6;
    float h = xr[0]*w[0] + xr[1]*w[1] + xr[2]*w[2] + xr[3]*w[3] + xr[4]*w[4] + xr[5]*w[5];
    s += h; s2 += h * h;
  }
  red[tid] = s; red[256 + tid] = s2;
  __syncthreads();
  for (int off = 128; off > 0; off >>= 1) {
    if (tid < off) { red[tid] += red[tid + off]; red[256 + tid] += red[256 + tid + off]; }
    __syncthreads();
  }
  if (tid == 0) {
    float inv_n = 1.0f / (float)ROWS;
    float mean = red[0] * inv_n;
    float var = red[256] * inv_n - mean * mean;
    snst[ch] = mean;
    snst[16 + ch] = rsqrtf(fmaxf(var, 0.f) + EPS_);
  }
}

// ---------------------------------------------------------------------------
// 3b. scorenet: h=relu(bn(xyz@w1)); s=h@w2+b; softmax+0.5 -> score (R,20,8)
// ---------------------------------------------------------------------------
__global__ __launch_bounds__(256) void k_sn_score(const float* __restrict__ xyz,
                                                  const float* __restrict__ w1,
                                                  const float* __restrict__ w2,
                                                  const float* __restrict__ bias,
                                                  const float* __restrict__ snst,
                                                  float* __restrict__ score) {
  __shared__ float sW1[96], sW2[128], sB[8], sM[16], sI[16];
  int tid = threadIdx.x;
  if (tid < 96)  sW1[tid] = w1[tid];
  if (tid < 128) sW2[tid] = w2[tid];
  if (tid < 8)   sB[tid] = bias[tid];
  if (tid < 16)  { sM[tid] = snst[tid]; sI[tid] = snst[16 + tid]; }
  __syncthreads();
  int t = blockIdx.x * 256 + tid;              // grid sized exactly: 2560 blocks
  const float* xr = xyz + (size_t)t * 6;
  float xv[6];
#pragma unroll
  for (int i = 0; i < 6; ++i) xv[i] = xr[i];
  float h[16];
#pragma unroll
  for (int ch = 0; ch < 16; ++ch) {
    float hv = 0.f;
#pragma unroll
    for (int i = 0; i < 6; ++i) hv += xv[i] * sW1[i * 16 + ch];
    hv = (hv - sM[ch]) * sI[ch];
    h[ch] = fmaxf(hv, 0.f);
  }
  float sv[8];
#pragma unroll
  for (int m = 0; m < 8; ++m) {
    float a = sB[m];
#pragma unroll
    for (int ch = 0; ch < 16; ++ch) a += h[ch] * sW2[ch * 8 + m];
    sv[m] = a;
  }
  float mx = sv[0];
#pragma unroll
  for (int m = 1; m < 8; ++m) mx = fmaxf(mx, sv[m]);
  float sum = 0.f;
#pragma unroll
  for (int m = 0; m < 8; ++m) { sv[m] = expf(sv[m] - mx); sum += sv[m]; }
  float inv = 1.0f / sum;
  float* o = score + (size_t)t * 8;
#pragma unroll
  for (int m = 0; m < 8; ++m) o[m] = sv[m] * inv + 0.5f;
}

// ---------------------------------------------------------------------------
// 4. PAConv layer 1 (C=3, O=64): small, VALU. One block (64 thr) per point.
// ---------------------------------------------------------------------------
__global__ __launch_bounds__(64) void k_paconv1(const float* __restrict__ score,
                                                const int* __restrict__ idx,
                                                const float* __restrict__ xT,
                                                const float* __restrict__ mat1,
                                                float* __restrict__ outPre) {
  __shared__ float sS[160], sT[24], sSum[8], sF[3];
  __shared__ int sIdx[20];
  int tid = threadIdx.x;
  int pt = blockIdx.x;
  int b = pt >> 10;
  for (int i = tid; i < 160; i += 64) sS[i] = score[(size_t)pt * 160 + i];
  if (tid < 20) sIdx[tid] = idx[(size_t)pt * 20 + tid];
  if (tid < 3)  sF[tid] = xT[(size_t)pt * 3 + tid];
  __syncthreads();
  if (tid < 24) {
    int c = tid >> 3, m = tid & 7;
    float acc = 0.f;
    for (int k = 0; k < 20; ++k)
      acc += sS[k * 8 + m] * xT[(size_t)(b * NN + sIdx[k]) * 3 + c];
    sT[c * 8 + m] = acc;
  }
  if (tid < 8) {
    float s = 0.f;
    for (int k = 0; k < 20; ++k) s += sS[k * 8 + tid];
    sSum[tid] = s;
  }
  __syncthreads();
  int o = tid;
  float acc = 0.f;
#pragma unroll
  for (int c = 0; c < 3; ++c) {
#pragma unroll
    for (int m = 0; m < 8; ++m) {
      float tv = sT[c * 8 + m];
      float v1 = tv - sSum[m] * sF[c];
      acc += v1 * mat1[c * 512 + m * 64 + o] + tv * mat1[(3 + c) * 512 + m * 64 + o];
    }
  }
  outPre[(size_t)pt * 64 + o] = acc;
}

// ---------------------------------------------------------------------------
// 5. Weight swizzle into WMMA B-fragment order.
//    Fragment (kt,nt): 32 lanes x 16 halves; col=nt*16+(lane&15), h=lane>>4;
//    k=kt*32+8h+e (e<8), k=kt*32+16+8h+(e-8) (e>=8).
// ---------------------------------------------------------------------------
__global__ void k_swizzle_paconv(const float* __restrict__ mat, _Float16* __restrict__ Wsw,
                                 int C, int O) {
  int t = blockIdx.x * blockDim.x + threadIdx.x;
  int total = 16 * C * O;
  if (t >= total) return;
  int NT = O >> 4;
  int frag = t >> 9, within = t & 511;
  int lane = within >> 4, e = within & 15;
  int kt = frag / NT, nt = frag - kt * NT;
  int k = kt * 32 + ((e & 8) << 1) + ((lane >> 4) << 3) + (e & 7);
  int col = nt * 16 + (lane & 15);
  int CM = C * 8;
  int half = k / CM, rr = k - half * CM;
  int c = rr >> 3, m = rr & 7;
  Wsw[t] = (_Float16)mat[(size_t)(half * C + c) * (8 * O) + m * O + col];
}

__global__ void k_swizzle_plain(const float* __restrict__ W, _Float16* __restrict__ Wsw,
                                int KD, int O) {
  int t = blockIdx.x * blockDim.x + threadIdx.x;
  if (t >= KD * O) return;
  int NT = O >> 4;
  int frag = t >> 9, within = t & 511;
  int lane = within >> 4, e = within & 15;
  int kt = frag / NT, nt = frag - kt * NT;
  int k = kt * 32 + ((e & 8) << 1) + ((lane >> 4) << 3) + (e & 7);
  int col = nt * 16 + (lane & 15);
  Wsw[t] = (_Float16)W[(size_t)k * O + col];
}

// ---------------------------------------------------------------------------
// 6. Fused PAConv GEMM (layers 2-4). 16-point tile per WG (128 thr / 4 waves).
//    Phase 1: build Fhat = [T - ssum*feat || T] (f16) in LDS, K-chunked (CH=64).
//    Phase 2: v_wmma_f32_16x16x32_f16 vs pre-swizzled weights; B loads use
//             static ping-pong buffers (no register rotation movs), one
//             k-step ahead of the WMMA chain.
// ---------------------------------------------------------------------------
template <int C, int O>
__global__ __launch_bounds__(128) void k_paconv_gemm(const float* __restrict__ score,
                                                     const int* __restrict__ idx,
                                                     const float* __restrict__ feat,
                                                     int featStride,
                                                     const _Float16* __restrict__ Wsw,
                                                     float* __restrict__ outPre) {
  constexpr int CH  = 64;          // channels per K-chunk
  constexpr int NCH = C / CH;      // 1 (layers 2,3) or 2 (layer 4)
  constexpr int KDL = 16 * CH;     // 1024 local K per chunk
  constexpr int LT  = KDL / 32;    // 32 k-steps per chunk (even)
  constexpr int NT  = O / 16;
  constexpr int TPW = NT / 4;      // N-tiles per wave
  extern __shared__ char smem[];
  float*    sS = (float*)smem;                       // 16*160 f32 = 10240 B
  _Float16* F  = (_Float16*)(smem + 16 * 160 * 4);   // 16*KDL f16 = 32768 B

  int tid = threadIdx.x;
  int ptBase = blockIdx.x * 16;
  for (int i = tid; i < 16 * 160; i += 128) sS[i] = score[(size_t)ptBase * 160 + i];

  int wave = tid >> 5, lane = tid & 31;
  int row = lane & 15;
  int koff = (lane < 16) ? 0 : 8;
  int rsel = (lane < 16) ? 0 : 8;
  int ntW = wave * TPW;

  v8f acc[TPW];
  v8f z = {0.f, 0.f, 0.f, 0.f, 0.f, 0.f, 0.f, 0.f};
#pragma unroll
  for (int t = 0; t < TPW; ++t) acc[t] = z;

  int grp = tid / CH;          // 0/1 -> two points in parallel (wave-uniform)
  int cl  = tid % CH;

  for (int ch = 0; ch < NCH; ++ch) {
    int c0 = ch * CH;
    __syncthreads();           // score ready (iter 0) / F free (iter >0)
    // ---- phase 1: build Fhat chunk ----
    for (int p = grp; p < 16; p += 2) {
      int pt = ptBase + p;
      int b = pt >> 10;
      const float* sp = sS + p * 160;
      float T[8], ss[8];
#pragma unroll
      for (int m = 0; m < 8; ++m) { T[m] = 0.f; ss[m] = 0.f; }
      for (int k = 0; k < 20; ++k) {
        int nb = idx[(size_t)pt * 20 + k];
        float f = feat[(size_t)(b * NN + nb) * featStride + c0 + cl];
#pragma unroll
        for (int m = 0; m < 8; ++m) { float s = sp[k * 8 + m]; T[m] += s * f; ss[m] += s; }
      }
      float fc = feat[(size_t)pt * featStride + c0 + cl];
#pragma unroll
      for (int m = 0; m < 8; ++m) {
        F[p * KDL + cl * 8 + m]          = (_Float16)(T[m] - ss[m] * fc);
        F[p * KDL + CH * 8 + cl * 8 + m] = (_Float16)T[m];
      }
    }
    __syncthreads();
    // ---- phase 2: ping-pong pipelined WMMA over this chunk's K range ----
    // global kt for local step lt:
    //   lt <  CH/4 : ktg = c0/4 + lt                (first half rows of Fhat)
    //   lt >= CH/4 : ktg = C/4 + c0/4 + (lt - CH/4) (second half rows)
    auto ktg_of = [&](int lt) {
      return (lt < CH / 4) ? (c0 / 4 + lt) : (C / 4 + c0 / 4 + (lt - CH / 4));
    };
    v16h bufA[TPW], bufB[TPW];
#pragma unroll
    for (int t = 0; t < TPW; ++t)
      bufA[t] = load_bfrag(Wsw, ktg_of(0) * NT + ntW + t, lane);

    for (int lt = 0; lt < LT; lt += 2) {
      // --- even step: consume bufA, load bufB (lt+1) ---
      union { v16h v; v8h h[2]; } a0;
      {
        const _Float16* fr = F + row * KDL + lt * 32 + koff;
        a0.h[0] = *(const v8h*)fr;
        a0.h[1] = *(const v8h*)(fr + 16);
      }
      int k1 = ktg_of(lt + 1);
#pragma unroll
      for (int t = 0; t < TPW; ++t) {
        const _Float16* bp = Wsw + ((size_t)(k1 * NT + ntW + t) * 32 + lane) * 16;
        __builtin_prefetch(bp + NT * 512, 0, 1);
        bufB[t] = *(const v16h*)bp;
      }
#pragma unroll
      for (int t = 0; t < TPW; ++t)
        acc[t] = __builtin_amdgcn_wmma_f32_16x16x32_f16(
            false, a0.v, false, bufA[t], (short)0, acc[t], false, false);
      // --- odd step: consume bufB, load bufA (lt+2, clamped) ---
      union { v16h v; v8h h[2]; } a1;
      {
        const _Float16* fr = F + row * KDL + (lt + 1) * 32 + koff;
        a1.h[0] = *(const v8h*)fr;
        a1.h[1] = *(const v8h*)(fr + 16);
      }
      int lt2 = (lt + 2 < LT) ? (lt + 2) : (lt + 1);
      int k2 = ktg_of(lt2);
#pragma unroll
      for (int t = 0; t < TPW; ++t) {
        const _Float16* bp = Wsw + ((size_t)(k2 * NT + ntW + t) * 32 + lane) * 16;
        __builtin_prefetch(bp + NT * 512, 0, 1);
        bufA[t] = *(const v16h*)bp;
      }
#pragma unroll
      for (int t = 0; t < TPW; ++t)
        acc[t] = __builtin_amdgcn_wmma_f32_16x16x32_f16(
            false, a1.v, false, bufB[t], (short)0, acc[t], false, false);
    }
  }
  // ---- store D (ISA C/D layout: VGPR r -> M = rsel + r, N = lane&15) ----
#pragma unroll
  for (int t = 0; t < TPW; ++t) {
    int col = (ntW + t) * 16 + (lane & 15);
#pragma unroll
    for (int r2 = 0; r2 < 8; ++r2)
      outPre[(size_t)(ptBase + rsel + r2) * O + col] = acc[t][r2];
  }
}

// ---------------------------------------------------------------------------
// 7. conv5 GEMM: (32768x512)@(512x1024), A staged fp32->fp16 in LDS.
//    grid (2048, 4); each WG: 16 rows x 256 cols (4 waves x 4 tiles);
//    B loads ping-pong pipelined one k-step ahead.
// ---------------------------------------------------------------------------
__global__ __launch_bounds__(128) void k_conv5_gemm(const float* __restrict__ A,
                                                    const _Float16* __restrict__ Wsw,
                                                    float* __restrict__ outPre) {
  constexpr int KD = 512, NTtot = 64, TPW = 4, O = 1024;
  constexpr int LT = KD / 32;                    // 16 k-steps (even)
  extern __shared__ char smem[];
  _Float16* F = (_Float16*)smem;                 // 16*512 f16 = 16 KB
  int tid = threadIdx.x;
  int ptBase = blockIdx.x * 16;
  int cg = blockIdx.y;
  for (int i = tid; i < 16 * KD; i += 128) F[i] = (_Float16)A[(size_t)ptBase * KD + i];
  __syncthreads();

  int wave = tid >> 5, lane = tid & 31;
  int row = lane & 15;
  int koff = (lane < 16) ? 0 : 8;
  int rsel = (lane < 16) ? 0 : 8;
  int ntBase = cg * 16 + wave * TPW;
  v8f acc[TPW];
  v8f z = {0.f, 0.f, 0.f, 0.f, 0.f, 0.f, 0.f, 0.f};
#pragma unroll
  for (int t = 0; t < TPW; ++t) acc[t] = z;

  v16h bufA[TPW], bufB[TPW];
#pragma unroll
  for (int t = 0; t < TPW; ++t) bufA[t] = load_bfrag(Wsw, ntBase + t, lane);

  for (int kt = 0; kt < LT; kt += 2) {
    // --- even step ---
    union { v16h v; v8h h[2]; } a0;
    {
      const _Float16* fr = F + row * KD + kt * 32 + koff;
      a0.h[0] = *(const v8h*)fr;
      a0.h[1] = *(const v8h*)(fr + 16);
    }
#pragma unroll
    for (int t = 0; t < TPW; ++t) {
      const _Float16* bp = Wsw + ((size_t)((kt + 1) * NTtot + ntBase + t) * 32 + lane) * 16;
      __builtin_prefetch(bp + NTtot * 512, 0, 1);
      bufB[t] = *(const v16h*)bp;
    }
#pragma unroll
    for (int t = 0; t < TPW; ++t)
      acc[t] = __builtin_amdgcn_wmma_f32_16x16x32_f16(
          false, a0.v, false, bufA[t], (short)0, acc[t], false, false);
    // --- odd step ---
    union { v16h v; v8h h[2]; } a1;
    {
      const _Float16* fr = F + row * KD + (kt + 1) * 32 + koff;
      a1.h[0] = *(const v8h*)fr;
      a1.h[1] = *(const v8h*)(fr + 16);
    }
    int kt2 = (kt + 2 < LT) ? (kt + 2) : (kt + 1);
#pragma unroll
    for (int t = 0; t < TPW; ++t) {
      const _Float16* bp = Wsw + ((size_t)(kt2 * NTtot + ntBase + t) * 32 + lane) * 16;
      __builtin_prefetch(bp + NTtot * 512, 0, 1);
      bufA[t] = *(const v16h*)bp;
    }
#pragma unroll
    for (int t = 0; t < TPW; ++t)
      acc[t] = __builtin_amdgcn_wmma_f32_16x16x32_f16(
          false, a1.v, false, bufB[t], (short)0, acc[t], false, false);
  }
#pragma unroll
  for (int t = 0; t < TPW; ++t) {
    int col = (ntBase + t) * 16 + (lane & 15);
#pragma unroll
    for (int r2 = 0; r2 < 8; ++r2)
      outPre[(size_t)(ptBase + rsel + r2) * O + col] = acc[t][r2];
  }
}

// ---------------------------------------------------------------------------
// 8. BN stats over rows (R x C); one block per column (deterministic).
// ---------------------------------------------------------------------------
__global__ __launch_bounds__(256) void k_bn_stats(const float* __restrict__ X, int R, int C,
                                                  float* __restrict__ mean,
                                                  float* __restrict__ inv) {
  __shared__ float red[512];
  int col = blockIdx.x, tid = threadIdx.x;
  float s = 0.f, s2 = 0.f;
  for (int r = tid; r < R; r += 256) {
    float v = X[(size_t)r * C + col];
    s += v; s2 += v * v;
  }
  red[tid] = s; red[256 + tid] = s2;
  __syncthreads();
  for (int off = 128; off > 0; off >>= 1) {
    if (tid < off) { red[tid] += red[tid + off]; red[256 + tid] += red[256 + tid + off]; }
    __syncthreads();
  }
  if (tid == 0) {
    float inv_n = 1.0f / (float)R;
    float mu = red[0] * inv_n;
    float var = red[256] * inv_n - mu * mu;
    mean[col] = mu;
    inv[col] = rsqrtf(fmaxf(var, 0.f) + EPS_);
  }
}

// 9. normalize+relu into pcat slot (row stride 512)
__global__ void k_bn_relu_to_pcat(const float* __restrict__ X, int C,
                                  const float* __restrict__ mean, const float* __restrict__ inv,
                                  float* __restrict__ pcat, int colOff) {
  int t = blockIdx.x * blockDim.x + threadIdx.x;
  if (t >= RR * C) return;
  int r = t / C, c = t - r * C;
  float v = (X[t] - mean[c]) * inv[c];
  pcat[(size_t)r * 512 + colOff + c] = fmaxf(v, 0.f);
}

// ---------------------------------------------------------------------------
// 10. fused normalize+relu+max/mean pool: out5Pre (R,1024) -> g (32,2048)
// ---------------------------------------------------------------------------
__global__ __launch_bounds__(256) void k_pool(const float* __restrict__ out5,
                                              const float* __restrict__ mean,
                                              const float* __restrict__ inv,
                                              float* __restrict__ g) {
  int b = blockIdx.x, tid = threadIdx.x;
  for (int c = tid; c < 1024; c += 256) {
    float mu = mean[c], iv = inv[c];
    float mx = -3.4e38f, sm = 0.f;
    for (int n = 0; n < NN; ++n) {
      float v = out5[((size_t)(b * NN + n)) * 1024 + c];
      v = fmaxf((v - mu) * iv, 0.f);
      mx = fmaxf(mx, v); sm += v;
    }
    g[(size_t)b * 2048 + c] = mx;
    g[(size_t)b * 2048 + 1024 + c] = sm * (1.0f / (float)NN);
  }
}

// 11. small matmul (32 x CI) @ (CI x CO); one thread per output
__global__ void k_mm(const float* __restrict__ A, const float* __restrict__ W,
                     float* __restrict__ Y, int CI, int CO) {
  int t = blockIdx.x * blockDim.x + threadIdx.x;
  if (t >= 32 * CO) return;
  int r = t / CO, c = t - r * CO;
  const float* a = A + (size_t)r * CI;
  float acc = 0.f;
  for (int j = 0; j < CI; ++j) acc += a[j] * W[(size_t)j * CO + c];
  Y[t] = acc;
}

// 12. BN over axis 0 (32 rows) + relu; one thread per column
__global__ void k_bn0_relu(const float* __restrict__ X, float* __restrict__ Y, int CO) {
  int c = blockIdx.x * blockDim.x + threadIdx.x;
  if (c >= CO) return;
  float s = 0.f, s2 = 0.f;
  for (int r = 0; r < 32; ++r) { float v = X[(size_t)r * CO + c]; s += v; s2 += v * v; }
  float mu = s * (1.0f / 32.0f);
  float var = s2 * (1.0f / 32.0f) - mu * mu;
  float iv = rsqrtf(fmaxf(var, 0.f) + EPS_);
  for (int r = 0; r < 32; ++r)
    Y[(size_t)r * CO + c] = fmaxf((X[(size_t)r * CO + c] - mu) * iv, 0.f);
}

// 13. final linear + bias -> d_out (32 x 40)
__global__ void k_lin3(const float* __restrict__ A, const float* __restrict__ W,
                       const float* __restrict__ bias, float* __restrict__ out) {
  int t = blockIdx.x * blockDim.x + threadIdx.x;
  if (t >= 32 * 40) return;
  int r = t / 40, c = t - r * 40;
  const float* a = A + (size_t)r * 256;
  float acc = bias[c];
  for (int j = 0; j < 256; ++j) acc += a[j] * W[(size_t)j * 40 + c];
  out[t] = acc;
}

// ---------------------------------------------------------------------------
extern "C" void kernel_launch(void* const* d_in, const int* in_sizes, int n_in,
                              void* d_out, int out_size, void* d_ws, size_t ws_size,
                              hipStream_t stream) {
  (void)in_sizes; (void)n_in; (void)out_size; (void)ws_size;
  // dict order: x, then per layer (mat, sn_w1, sn_w2, sn_b), then conv5/lin*
  const float* x     = (const float*)d_in[0];
  const float* mats[4]  = { (const float*)d_in[1], (const float*)d_in[5],
                            (const float*)d_in[9], (const float*)d_in[13] };
  const float* snw1[4]  = { (const float*)d_in[2], (const float*)d_in[6],
                            (const float*)d_in[10], (const float*)d_in[14] };
  const float* snw2[4]  = { (const float*)d_in[3], (const float*)d_in[7],
                            (const float*)d_in[11], (const float*)d_in[15] };
  const float* snb[4]   = { (const float*)d_in[4], (const float*)d_in[8],
                            (const float*)d_in[12], (const float*)d_in[16] };
  const float* conv5_w = (const float*)d_in[17];
  const float* lin1_w  = (const float*)d_in[18];
  const float* lin2_w  = (const float*)d_in[19];
  const float* lin3_w  = (const float*)d_in[20];
  const float* lin3_b  = (const float*)d_in[21];

  char* ws = (char*)d_ws;
  size_t off = 0;
  auto carve = [&](size_t bytes) { char* p = ws + off; off += (bytes + 255) & ~(size_t)255; return p; };
  float*    xT     = (float*)carve((size_t)RR * 3 * 4);
  int*      idxB   = (int*)  carve((size_t)RR * 20 * 4);
  float*    xyz    = (float*)carve((size_t)RR * 20 * 6 * 4);
  float*    score  = (float*)carve((size_t)RR * 20 * 8 * 4);
  float*    pcat   = (float*)carve((size_t)RR * 512 * 4);
  float*    outPre = (float*)carve((size_t)RR * 256 * 4);
  float*    out5   = (float*)carve((size_t)RR * 1024 * 4);
  _Float16* Wsw    = (_Float16*)carve((size_t)1 << 20);
  float*    stats  = (float*)carve(2048 * 4);   // mean [0,1024), inv [1024,2048)
  float*    snst   = (float*)carve(32 * 4);
  float*    gbuf   = (float*)carve(32 * 2048 * 4);
  float*    h1     = (float*)carve(32 * 512 * 4);
  float*    h1n    = (float*)carve(32 * 512 * 4);
  float*    h2     = (float*)carve(32 * 256 * 4);
  float*    h2n    = (float*)carve(32 * 256 * 4);
  float* meanP = stats;
  float* invP  = stats + 1024;

  // geometry
  k_transpose_x<<<(RR * 3 + 255) / 256, 256, 0, stream>>>(x, xT);
  k_knn<<<RR / 256, 256, 0, stream>>>(x, idxB);
  k_xyz<<<(RR * 20) / 256, 256, 0, stream>>>(xT, idxB, xyz);

  const int   Cout[4]   = { 64, 64, 128, 256 };
  const int   colOff[4] = { 0, 64, 128, 256 };
  const size_t shPaconv = 16 * 160 * 4 + 16 * 1024 * 2;   // 43008 B

  for (int l = 0; l < 4; ++l) {
    // scorenet
    k_sn_stats<<<16, 256, 0, stream>>>(xyz, snw1[l], snst);
    k_sn_score<<<(RR * 20) / 256, 256, 0, stream>>>(xyz, snw1[l], snw2[l], snb[l], snst, score);
    // paconv
    if (l == 0) {
      k_paconv1<<<RR, 64, 0, stream>>>(score, idxB, xT, mats[0], outPre);
    } else {
      // feat for layer l lives at pcat cols: l=1 -> [0,64), l=2 -> [64,128), l=3 -> [128,256)
      const float* feat = pcat + (l == 1 ? 0 : (l == 2 ? 64 : 128));
      if (l == 1) {
        k_swizzle_paconv<<<(16 * 64 * 64 + 255) / 256, 256, 0, stream>>>(mats[1], Wsw, 64, 64);
        k_paconv_gemm<64, 64><<<RR / 16, 128, shPaconv, stream>>>(score, idxB, feat, 512, Wsw, outPre);
      } else if (l == 2) {
        k_swizzle_paconv<<<(16 * 64 * 128 + 255) / 256, 256, 0, stream>>>(mats[2], Wsw, 64, 128);
        k_paconv_gemm<64, 128><<<RR / 16, 128, shPaconv, stream>>>(score, idxB, feat, 512, Wsw, outPre);
      } else {
        k_swizzle_paconv<<<(16 * 128 * 256 + 255) / 256, 256, 0, stream>>>(mats[3], Wsw, 128, 256);
        k_paconv_gemm<128, 256><<<RR / 16, 128, shPaconv, stream>>>(score, idxB, feat, 512, Wsw, outPre);
      }
    }
    // BN + relu into pcat
    int O = Cout[l];
    k_bn_stats<<<O, 256, 0, stream>>>(outPre, RR, O, meanP, invP);
    k_bn_relu_to_pcat<<<(RR * O + 255) / 256, 256, 0, stream>>>(outPre, O, meanP, invP, pcat, colOff[l]);
  }

  // conv5
  k_swizzle_plain<<<(512 * 1024 + 255) / 256, 256, 0, stream>>>(conv5_w, Wsw, 512, 1024);
  k_conv5_gemm<<<dim3(RR / 16, 4), 128, 16 * 512 * 2, stream>>>(pcat, Wsw, out5);
  k_bn_stats<<<1024, 256, 0, stream>>>(out5, RR, 1024, meanP, invP);
  k_pool<<<BB, 256, 0, stream>>>(out5, meanP, invP, gbuf);

  // head
  k_mm<<<(32 * 512 + 255) / 256, 256, 0, stream>>>(gbuf, lin1_w, h1, 2048, 512);
  k_bn0_relu<<<(512 + 255) / 256, 256, 0, stream>>>(h1, h1n, 512);
  k_mm<<<(32 * 256 + 255) / 256, 256, 0, stream>>>(h1n, lin2_w, h2, 512, 256);
  k_bn0_relu<<<1, 256, 0, stream>>>(h2, h2n, 256);
  k_lin3<<<(32 * 40 + 255) / 256, 256, 0, stream>>>(h2n, lin3_w, lin3_b, (float*)d_out);
}
